// DecoderWithoutAttention_45629732553082
// MI455X (gfx1250) — compile-verified, matrix operand order checked by reading, not run on hardware
//
#include <hip/hip_runtime.h>
#include <stddef.h>

// Problem constants (from reference)
#define BB   64
#define TT   49
#define EE   512
#define HH   512
#define ENCC 1024
#define VV   32000
#define G4H  (4 * HH)   // 2048
#define XLD  (TT * EE)  // row stride (elements) between batches in embeddings

typedef __attribute__((ext_vector_type(16))) __bf16       v16bf;
typedef __attribute__((ext_vector_type(8)))  float        v8f;
typedef __attribute__((ext_vector_type(4)))  unsigned int u32x4;

union TileBF { v16bf v; u32x4 q[2]; };

// ---------------------------------------------------------------------------
// fp32 -> bf16 round-to-nearest-even
__device__ inline unsigned short f32_to_bf16(float f) {
    unsigned int u = __float_as_uint(f);
    unsigned int lsb = (u >> 16) & 1u;
    u += 0x7fffu + lsb;
    return (unsigned short)(u >> 16);
}

// 16x32 bf16 WMMA tile from lane base + compile-time element offset
// (one b128 pair with immediate offsets)
__device__ inline v16bf tile_at(const unsigned short* __restrict__ lane_base, int elem_off) {
    TileBF t;
    t.q[0] = *(const u32x4*)(lane_base + elem_off);
    t.q[1] = *(const u32x4*)(lane_base + elem_off + 16);
    return t.v;
}

__device__ inline v8f wmma_bf16(v16bf a, v16bf b, v8f c) {
    return __builtin_amdgcn_wmma_f32_16x16x32_bf16(false, a, false, b, (short)0, c, false, false);
}

// ---------------------------------------------------------------------------
__global__ void cvt_bf16_kernel(const float* __restrict__ src,
                                unsigned short* __restrict__ dst, size_t n) {
    size_t i = (size_t)blockIdx.x * blockDim.x + threadIdx.x;
    size_t stride = (size_t)gridDim.x * blockDim.x;
    for (; i < n; i += stride) dst[i] = f32_to_bf16(src[i]);
}

__global__ void bias_sum_kernel(const float* __restrict__ b_ih,
                                const float* __restrict__ b_hh,
                                float* __restrict__ bias_sum) {
    int i = blockIdx.x * blockDim.x + threadIdx.x;
    if (i < G4H) bias_sum[i] = b_ih[i] + b_hh[i];
}

// h0 = enc @ W_init_h^T + b ; c0 likewise (tiny: 134 MFLOP, fp32 for accuracy)
__global__ void init_state_kernel(const float* __restrict__ enc,
                                  const float* __restrict__ Wh,
                                  const float* __restrict__ bh,
                                  const float* __restrict__ Wc,
                                  const float* __restrict__ bc,
                                  float* __restrict__ h, float* __restrict__ c,
                                  unsigned short* __restrict__ h_bf) {
    int idx = blockIdx.x * blockDim.x + threadIdx.x;   // 0 .. B*H-1
    if (idx >= BB * HH) return;
    int b = idx / HH, j = idx % HH;
    const float* e  = enc + (size_t)b * ENCC;
    const float* wh = Wh  + (size_t)j * ENCC;
    const float* wc = Wc  + (size_t)j * ENCC;
    float sh = bh[j], sc = bc[j];
    for (int k = 0; k < ENCC; ++k) { sh = fmaf(e[k], wh[k], sh); sc = fmaf(e[k], wc[k], sc); }
    h[idx] = sh; c[idx] = sc; h_bf[idx] = f32_to_bf16(sh);
}

// ---------------------------------------------------------------------------
// gates(64x2048) = x_t @ W_ih^T + h @ W_hh^T + (b_ih + b_hh)
// one wave per 16-col N-tile (128 waves); A and B double-buffered across
// k-steps so WMMA groups overlap the next group's loads.
__global__ __launch_bounds__(256) void gates_kernel(
        const unsigned short* __restrict__ x_bf,   // emb_bf + t*EE
        const unsigned short* __restrict__ h_bf,
        const unsigned short* __restrict__ Wih,
        const unsigned short* __restrict__ Whh,
        const float* __restrict__ bias_sum,
        float* __restrict__ gates) {
    const int lane = threadIdx.x & 31;
    const int wave = (int)((blockIdx.x * blockDim.x + threadIdx.x) >> 5);
    const int n0   = wave * 16;
    const int half = lane >> 4;
    const int r    = lane & 15;

    const unsigned short* bih = Wih  + (size_t)(n0 + r) * EE  + half * 8;
    const unsigned short* bhh = Whh  + (size_t)(n0 + r) * HH  + half * 8;
    const unsigned short* ax  = x_bf + (size_t)r * XLD        + half * 8;
    const unsigned short* ah  = h_bf + (size_t)r * HH         + half * 8;

    v8f acc[4] = {};
    // ---- pass 1: x @ W_ih^T (double-buffered)
    {
        v16bf bm = tile_at(bih, 0);
        v16bf am[4];
#pragma unroll
        for (int mt = 0; mt < 4; ++mt) am[mt] = tile_at(ax, mt * 16 * XLD);
#pragma unroll
        for (int k0 = 0; k0 < EE; k0 += 32) {
            const int kn = k0 + 32;
            v16bf bn {};
            v16bf an[4] {};
            if (kn < EE) {
                bn = tile_at(bih, kn);
#pragma unroll
                for (int mt = 0; mt < 4; ++mt) an[mt] = tile_at(ax, mt * 16 * XLD + kn);
            }
#pragma unroll
            for (int mt = 0; mt < 4; ++mt) acc[mt] = wmma_bf16(am[mt], bm, acc[mt]);
            bm = bn;
#pragma unroll
            for (int mt = 0; mt < 4; ++mt) am[mt] = an[mt];
        }
    }
    // ---- pass 2: h @ W_hh^T (double-buffered)
    {
        v16bf bm = tile_at(bhh, 0);
        v16bf am[4];
#pragma unroll
        for (int mt = 0; mt < 4; ++mt) am[mt] = tile_at(ah, mt * 16 * HH);
#pragma unroll
        for (int k0 = 0; k0 < HH; k0 += 32) {
            const int kn = k0 + 32;
            v16bf bn {};
            v16bf an[4] {};
            if (kn < HH) {
                bn = tile_at(bhh, kn);
#pragma unroll
                for (int mt = 0; mt < 4; ++mt) an[mt] = tile_at(ah, mt * 16 * HH + kn);
            }
#pragma unroll
            for (int mt = 0; mt < 4; ++mt) acc[mt] = wmma_bf16(am[mt], bm, acc[mt]);
            bm = bn;
#pragma unroll
            for (int mt = 0; mt < 4; ++mt) am[mt] = an[mt];
        }
    }
    const int n = n0 + r;
    const float bias = bias_sum[n];
#pragma unroll
    for (int mt = 0; mt < 4; ++mt)
#pragma unroll
        for (int rr = 0; rr < 8; ++rr) {
            int m = mt * 16 + half * 8 + rr;
            gates[(size_t)m * G4H + n] = acc[mt][rr] + bias;
        }
}

// elementwise LSTM cell + mask select
__global__ void cell_kernel(const float* __restrict__ gates,
                            const int* __restrict__ cap_len,
                            float* __restrict__ h, float* __restrict__ c,
                            unsigned short* __restrict__ h_bf, int t) {
    int idx = blockIdx.x * blockDim.x + threadIdx.x;   // 0 .. B*H-1
    if (idx >= BB * HH) return;
    int b = idx / HH, j = idx % HH;
    const float* g = gates + (size_t)b * G4H;
    float ig = 1.0f / (1.0f + __expf(-g[j]));
    float fg = 1.0f / (1.0f + __expf(-g[HH + j]));
    float gg = tanhf(g[2 * HH + j]);
    float og = 1.0f / (1.0f + __expf(-g[3 * HH + j]));
    float cn = fg * c[idx] + ig * gg;
    float hn = og * tanhf(cn);
    bool on = t < (cap_len[b] - 1);
    float hs = on ? hn : h[idx];
    float cs = on ? cn : c[idx];
    h[idx] = hs; c[idx] = cs; h_bf[idx] = f32_to_bf16(hs);
}

// preds(64x32000) = h @ W_fc^T + b_fc, masked rows -> 0.
// A (64x512 bf16, 64KB) staged once per block into LDS, shared by 8 waves.
// Each wave owns TWO 16-col N-tiles: per k-step, 2 global B-pairs + 4 LDS
// A-pairs feed 8 WMMAs; B double-buffered across k-steps.
__global__ __launch_bounds__(256) void fc_kernel(
        const unsigned short* __restrict__ h_bf,
        const unsigned short* __restrict__ Wfc,
        const float* __restrict__ b_fc,
        const int* __restrict__ cap_len,
        float* __restrict__ out, int t) {
    __shared__ __align__(16) unsigned short Alds[BB * HH];   // 64 KB

    {   // cooperative stage: 4096 x 16B, 256 threads -> 16 vectors each
        const u32x4* src = (const u32x4*)h_bf;
        u32x4* dst = (u32x4*)Alds;
        int tid = threadIdx.x;
#pragma unroll
        for (int i = 0; i < (BB * HH / 8) / 256; ++i)
            dst[tid + i * 256] = src[tid + i * 256];
    }
    __syncthreads();

    const int lane = threadIdx.x & 31;
    const int wave = (int)((blockIdx.x * blockDim.x + threadIdx.x) >> 5);
    const int n0   = wave * 32;                 // two N-tiles per wave
    const int half = lane >> 4;
    const int r    = lane & 15;

    const unsigned short* bbase0 = Wfc + (size_t)(n0 + r) * HH + half * 8;
    const unsigned short* bbase1 = Wfc + (size_t)(n0 + 16 + r) * HH + half * 8;
    const unsigned short* abase  = Alds + r * HH + half * 8;   // LDS, imm offsets

    v8f acc0[4] = {}, acc1[4] = {};
    v16bf b0 = tile_at(bbase0, 0);
    v16bf b1 = tile_at(bbase1, 0);
#pragma unroll
    for (int k0 = 0; k0 < HH; k0 += 32) {
        const int kn = k0 + 32;
        v16bf b0n {}, b1n {};
        if (kn < HH) { b0n = tile_at(bbase0, kn); b1n = tile_at(bbase1, kn); }
        v16bf am[4];
#pragma unroll
        for (int mt = 0; mt < 4; ++mt) am[mt] = tile_at(abase, mt * 16 * HH + k0);
#pragma unroll
        for (int mt = 0; mt < 4; ++mt) {
            acc0[mt] = wmma_bf16(am[mt], b0, acc0[mt]);
            acc1[mt] = wmma_bf16(am[mt], b1, acc1[mt]);
        }
        b0 = b0n; b1 = b1n;
    }
    const int n = n0 + r;
    const float bias0 = b_fc[n];
    const float bias1 = b_fc[n + 16];
#pragma unroll
    for (int mt = 0; mt < 4; ++mt)
#pragma unroll
        for (int rr = 0; rr < 8; ++rr) {
            int m = mt * 16 + half * 8 + rr;
            bool on = t < (cap_len[m] - 1);
            size_t rowoff = ((size_t)m * TT + t) * VV;
            out[rowoff + n]      = on ? (acc0[mt][rr] + bias0) : 0.0f;
            out[rowoff + n + 16] = on ? (acc1[mt][rr] + bias1) : 0.0f;
        }
}

// decode_lengths appended after predictions (as float, per output dtype)
__global__ void tail_kernel(const int* __restrict__ cap_len, float* __restrict__ out) {
    int b = threadIdx.x;
    if (b < BB) out[(size_t)BB * TT * VV + b] = (float)(cap_len[b] - 1);
}

// ---------------------------------------------------------------------------
extern "C" void kernel_launch(void* const* d_in, const int* in_sizes, int n_in,
                              void* d_out, int out_size, void* d_ws, size_t ws_size,
                              hipStream_t stream) {
    (void)in_sizes; (void)n_in; (void)out_size; (void)ws_size;
    const float* emb      = (const float*)d_in[0];
    const float* enc      = (const float*)d_in[1];
    const int*   cap_len  = (const int*)  d_in[2];
    const float* W_ih     = (const float*)d_in[3];
    const float* W_hh     = (const float*)d_in[4];
    const float* b_ih     = (const float*)d_in[5];
    const float* b_hh     = (const float*)d_in[6];
    const float* W_init_h = (const float*)d_in[7];
    const float* b_init_h = (const float*)d_in[8];
    const float* W_init_c = (const float*)d_in[9];
    const float* b_init_c = (const float*)d_in[10];
    const float* W_fc     = (const float*)d_in[11];
    const float* b_fc     = (const float*)d_in[12];
    float* out = (float*)d_out;

    // workspace layout (256B aligned offsets)
    char* ws = (char*)d_ws;
    size_t off = 0;
    auto take = [&](size_t bytes) { size_t r = off; off = (off + bytes + 255) & ~(size_t)255; return r; };
    unsigned short* Wih_bf  = (unsigned short*)(ws + take((size_t)G4H * EE * 2));
    unsigned short* Whh_bf  = (unsigned short*)(ws + take((size_t)G4H * HH * 2));
    unsigned short* Wfc_bf  = (unsigned short*)(ws + take((size_t)VV * HH * 2));
    unsigned short* emb_bf  = (unsigned short*)(ws + take((size_t)BB * TT * EE * 2));
    float*          biassum = (float*)         (ws + take((size_t)G4H * 4));
    float*          h_f32   = (float*)         (ws + take((size_t)BB * HH * 4));
    float*          c_f32   = (float*)         (ws + take((size_t)BB * HH * 4));
    unsigned short* h_bf    = (unsigned short*)(ws + take((size_t)BB * HH * 2));
    float*          gates   = (float*)         (ws + take((size_t)BB * G4H * 4));

    const int TB = 256;
    auto cvt = [&](const float* s, unsigned short* d, size_t n) {
        int blocks = (int)((n + TB - 1) / TB);
        if (blocks > 16384) blocks = 16384;
        cvt_bf16_kernel<<<blocks, TB, 0, stream>>>(s, d, n);
    };
    // one-time conversions
    cvt(W_ih, Wih_bf, (size_t)G4H * EE);
    cvt(W_hh, Whh_bf, (size_t)G4H * HH);
    cvt(W_fc, Wfc_bf, (size_t)VV * HH);
    cvt(emb,  emb_bf, (size_t)BB * TT * EE);
    bias_sum_kernel<<<(G4H + TB - 1) / TB, TB, 0, stream>>>(b_ih, b_hh, biassum);
    init_state_kernel<<<(BB * HH + TB - 1) / TB, TB, 0, stream>>>(
        enc, W_init_h, b_init_h, W_init_c, b_init_c, h_f32, c_f32, h_bf);

    const int gates_blocks = (G4H / 16) * 32 / TB;      // 128 waves -> 16 blocks
    const int fc_blocks    = (VV / 32) * 32 / TB;       // 1000 waves -> 125 blocks
    for (int t = 0; t < TT; ++t) {
        const unsigned short* x_t = emb_bf + (size_t)t * EE;
        gates_kernel<<<gates_blocks, TB, 0, stream>>>(
            x_t, h_bf, Wih_bf, Whh_bf, biassum, gates);
        cell_kernel<<<(BB * HH + TB - 1) / TB, TB, 0, stream>>>(
            gates, cap_len, h_f32, c_f32, h_bf, t);
        fc_kernel<<<fc_blocks, TB, 0, stream>>>(
            h_bf, Wfc_bf, b_fc, cap_len, out, t);
    }
    tail_kernel<<<1, 64, 0, stream>>>(cap_len, out);
}
